// SynthesisBlock_74947179316049
// MI455X (gfx1250) — matrix operand order repaired and checked by reading, not compiled, not run on hardware
//
#include <hip/hip_runtime.h>
#include <hip/hip_bf16.h>

// ---------------------------------------------------------------------------
// StyleGAN2 SynthesisBlock for MI455X (gfx1250), wave32 + WMMA bf16 +
// async global->LDS staging (ASYNCcnt path).
//
// Algebraic reductions vs the reference:
//  * second modulated conv is dead code (result discarded) -> skipped
//  * modulate (s) and demodulate (d) are both per-(b,cout) scalars ->
//    one SHARED 3x3 conv + per-(b,co) scale t0 = s * rsqrt(s^2*Q + 1e-8)
// ---------------------------------------------------------------------------

typedef __attribute__((ext_vector_type(8)))  float  v8f;
typedef __attribute__((ext_vector_type(16))) __bf16 v16bf;
typedef __attribute__((ext_vector_type(4)))  int    v4i;

#define Bn   8
#define H0   32
#define H1   64
#define CIN  512
#define COUT 512

// halo'd upsampled activation: [B][66][66][512] bf16 (1px zero halo)
#define XH_W   66
#define XH_ROW (XH_W * CIN)      // elements per y-row
#define XH_IMG (XH_W * XH_ROW)   // elements per image

#define LDS_STRIDE 80            // bytes per 32-bf16 row (64B + 16B pad: conflict-free)
#define TILE_BYTES (128 * LDS_STRIDE)

#if defined(__gfx1250__) && __has_builtin(__builtin_amdgcn_global_load_async_to_lds_b128)
#define USE_ASYNC 1
#else
#define USE_ASYNC 0
#endif

#if USE_ASYNC
typedef __attribute__((address_space(1))) v4i* gv4i_p;   // global int4*
typedef __attribute__((address_space(3))) v4i* lv4i_p;   // LDS int4*

__device__ __forceinline__ void async_cp16(const void* g, void* l) {
  __builtin_amdgcn_global_load_async_to_lds_b128((gv4i_p)(g), (lv4i_p)(l), 0, 0);
}
#if __has_builtin(__builtin_amdgcn_s_wait_asynccnt)
#define WAIT_ASYNC(n) __builtin_amdgcn_s_wait_asynccnt(n)
#else
#define WAIT_ASYNC(n) asm volatile("s_wait_asynccnt %0" ::"i"(n) : "memory")
#endif
#endif

union FragU { uint4 u[2]; v16bf v; };

// ---------------------------------------------------------------------------
// 1) bilinear 2x upsample (half-pixel centers, clamped edges) -> bf16 + halo
// ---------------------------------------------------------------------------
__global__ __launch_bounds__(256) void k_upsample(const float* __restrict__ x,
                                                  __bf16* __restrict__ xh) {
  int t = blockIdx.x * 256 + threadIdx.x;          // [B][66][66][64 c-blocks of 8]
  int c8 = t & 63;
  int xx = (t >> 6) % 66;
  int yy = ((t >> 6) / 66) % 66;
  int b  = t / (64 * 66 * 66);

  union { __bf16 h[8]; uint4 u; } pk;
  size_t off = (size_t)b * XH_IMG + (size_t)yy * XH_ROW + (size_t)xx * CIN + c8 * 8;

  if (yy == 0 || yy == 65 || xx == 0 || xx == 65) {
    pk.u = make_uint4(0u, 0u, 0u, 0u);             // zero halo (SAME padding)
  } else {
    int oy = yy - 1, ox = xx - 1;
    // out coord i maps to input i/2 - 0.25 -> even: {0.25@j-1, 0.75@j}, odd: {0.75@j, 0.25@j+1}
    int iy0, iy1, ix0, ix1; float wy0, wy1, wx0, wx1;
    int jy = oy >> 1, jx = ox >> 1;
    if (oy & 1) { iy0 = jy;     iy1 = jy + 1; wy0 = 0.75f; wy1 = 0.25f; }
    else        { iy0 = jy - 1; iy1 = jy;     wy0 = 0.25f; wy1 = 0.75f; }
    if (ox & 1) { ix0 = jx;     ix1 = jx + 1; wx0 = 0.75f; wx1 = 0.25f; }
    else        { ix0 = jx - 1; ix1 = jx;     wx0 = 0.25f; wx1 = 0.75f; }
    iy0 = max(iy0, 0); iy1 = min(iy1, H0 - 1);
    ix0 = max(ix0, 0); ix1 = min(ix1, H0 - 1);
    const float* p00 = x + (((size_t)(b * H0 + iy0) * H0 + ix0) * CIN) + c8 * 8;
    const float* p01 = x + (((size_t)(b * H0 + iy0) * H0 + ix1) * CIN) + c8 * 8;
    const float* p10 = x + (((size_t)(b * H0 + iy1) * H0 + ix0) * CIN) + c8 * 8;
    const float* p11 = x + (((size_t)(b * H0 + iy1) * H0 + ix1) * CIN) + c8 * 8;
#pragma unroll
    for (int i = 0; i < 8; ++i) {
      float v = wy0 * (wx0 * p00[i] + wx1 * p01[i]) +
                wy1 * (wx0 * p10[i] + wx1 * p11[i]);
      pk.h[i] = (__bf16)v;
    }
  }
  *(uint4*)(xh + off) = pk.u;
}

// ---------------------------------------------------------------------------
// 2) weights: [3][3][Cin][Cout] f32 -> wT[tap][co][ci] bf16 (ci contiguous)
// ---------------------------------------------------------------------------
__global__ __launch_bounds__(256) void k_wconv(const float* __restrict__ conv0,
                                               __bf16* __restrict__ wT) {
  int t = blockIdx.x * 256 + threadIdx.x;          // [9][512][64 ci-blocks of 8]
  int p   = t & 63;
  int co  = (t >> 6) & 511;
  int tap = t >> 15;
  int ci = p * 8;
  union { __bf16 h[8]; uint4 u; } pk;
#pragma unroll
  for (int i = 0; i < 8; ++i)
    pk.h[i] = (__bf16)conv0[((size_t)(tap * CIN + ci + i)) * COUT + co];
  *(uint4*)(wT + ((size_t)(tap * COUT + co)) * CIN + ci) = pk.u;
}

// ---------------------------------------------------------------------------
// 3) Q[co] = sum_{tap,ci} conv0^2    (coalesced over co)
// ---------------------------------------------------------------------------
__global__ void k_q(const float* __restrict__ conv0, float* __restrict__ Q) {
  int co = blockIdx.x * 256 + threadIdx.x;
  if (co >= COUT) return;
  float q = 0.f;
  for (int i = 0; i < 9 * CIN; ++i) { float v = conv0[(size_t)i * COUT + co]; q += v * v; }
  Q[co] = q;
}

// ---------------------------------------------------------------------------
// 4) t0[b,co] = s * rsqrt(s^2*Q + 1e-8),   s = w[b,:]@y0_kernel[:,co] + y0_bias
// ---------------------------------------------------------------------------
__global__ void k_style(const float* __restrict__ w, const float* __restrict__ yk,
                        const float* __restrict__ yb, const float* __restrict__ Q,
                        float* __restrict__ t0) {
  int g = blockIdx.x * 256 + threadIdx.x;          // B*COUT = 4096
  if (g >= Bn * COUT) return;
  int b = g >> 9, co = g & 511;
  float s = yb[co];
  for (int k = 0; k < 512; ++k) s += w[b * 512 + k] * yk[(size_t)k * COUT + co];
  float d = rsqrtf(s * s * Q[co] + 1e-8f);
  t0[g] = s * d;
}

// ---------------------------------------------------------------------------
// 5) main implicit-GEMM conv + epilogue.
//    Block tile: M=128 px (2 output rows), N=128 cout. 8 waves, each 32x64.
//    K loop: 9 taps x 16 cin-chunks of 32 -> 144 iters, 8 WMMAs/wave/iter.
//    Staging: async global->LDS (double-buffered), ASYNCcnt-tracked.
// ---------------------------------------------------------------------------
__global__ __launch_bounds__(256) void k_conv(
    const __bf16* __restrict__ xh, const __bf16* __restrict__ wT,
    const float* __restrict__ t0, const float* __restrict__ noise,
    const float* __restrict__ ns0, const float* __restrict__ ns1,
    const float* __restrict__ bias0, const float* __restrict__ bias1,
    float* __restrict__ out) {
#if USE_ASYNC
  __shared__ __align__(16) char ldsA[2 * TILE_BYTES];
  __shared__ __align__(16) char ldsB[2 * TILE_BYTES];
#else
  __shared__ __align__(16) char ldsA[TILE_BYTES];
  __shared__ __align__(16) char ldsB[TILE_BYTES];
#endif

  const int tid  = threadIdx.x;
  const int nblk = blockIdx.x;   // cout block (0..3)
  const int y2   = blockIdx.y;   // output row-pair (0..31)
  const int b    = blockIdx.z;   // batch (0..7)

  const int lane = tid & 31;
  const int wid  = tid >> 5;
  const int wm   = wid & 3;      // 4-way M split (32 rows each)
  const int wn   = wid >> 2;     // 2-way N split (64 cols each)

  const int srow   = tid >> 2;   // staging: row 0..63 (and +64)
  const int spiece = tid & 3;    // staging: 16B piece

  v8f acc[2][4];
  const v8f vzero = {0.f, 0.f, 0.f, 0.f, 0.f, 0.f, 0.f, 0.f};
#pragma unroll
  for (int mi = 0; mi < 2; ++mi)
#pragma unroll
    for (int ni = 0; ni < 4; ++ni) acc[mi][ni] = vzero;

  // per-iteration global source addresses for this thread's 4 pieces
  auto srcAddrs = [&](int it, const __bf16*& a0, const __bf16*& a1,
                      const __bf16*& b0p, const __bf16*& b1p) {
    int tap = it >> 4;                 // 0..8
    int kc  = it & 15;                 // cin chunk of 32
    int dy = tap / 3, dx = tap - dy * 3;
    int iy = 2 * y2 + dy;              // halo'd row index (m=srow -> ry=2*y2)
    int ix = srow + dx;
    size_t abase = (size_t)b * XH_IMG + (size_t)iy * XH_ROW + (size_t)ix * CIN
                 + kc * 32 + spiece * 8;
    a0 = xh + abase;
    a1 = xh + abase + XH_ROW;          // m = srow+64 -> ry = 2*y2+1
    int co = nblk * 128 + srow;
    size_t bbase = ((size_t)(tap * COUT + co)) * CIN + kc * 32 + spiece * 8;
    b0p = wT + bbase;
    b1p = wT + bbase + (size_t)64 * CIN;
  };

  // A frag (16x32 bf16): lane<16: v0-3=K0..7,v4-7=K16..23 ; lane>=16: K8..15,K24..31
  const int foA = (lane & 16);        // byte offset 0/16, second read +32
  // B frag (32x16 bf16): lane<16: v0-3=K0..7,v4-7=K16..23 ; lane>=16: K8..15,K24..31
  const int foB = (lane & 16);        // byte offset 0/16, second read +32
  const int mrow0 = wm * 32 + (lane & 15);
  const int nrow0 = wn * 64 + (lane & 15);

#if USE_ASYNC
  // -------- async global->LDS double-buffered pipeline --------
  char* stA = ldsA + srow * LDS_STRIDE + spiece * 16;
  char* stB = ldsB + srow * LDS_STRIDE + spiece * 16;

  auto issue = [&](int it, int buf) {
    const __bf16 *a0, *a1, *b0p, *b1p;
    srcAddrs(it, a0, a1, b0p, b1p);
    char* dA = stA + buf * TILE_BYTES;
    char* dB = stB + buf * TILE_BYTES;
    async_cp16(a0,  dA);
    async_cp16(a1,  dA + 64 * LDS_STRIDE);
    async_cp16(b0p, dB);
    async_cp16(b1p, dB + 64 * LDS_STRIDE);
  };

  issue(0, 0);
  for (int it = 0; it < 144; ++it) {
    const int cur = it & 1;
    if (it + 1 < 144) {
      issue(it + 1, cur ^ 1);   // buf[cur^1] free: last read finished before
                                // the trailing barrier of iter it-1
      WAIT_ASYNC(4);            // in-order completion: <=4 left => cur's 4 done
    } else {
      WAIT_ASYNC(0);
    }
    __syncthreads();            // all waves' async fills of buf[cur] visible

    const char* bA = ldsA + cur * TILE_BYTES;
    const char* bB = ldsB + cur * TILE_BYTES;
    FragU fa[2], fb[4];
#pragma unroll
    for (int mi = 0; mi < 2; ++mi) {
      const char* p = bA + (mrow0 + mi * 16) * LDS_STRIDE + foA;
      fa[mi].u[0] = *(const uint4*)p;
      fa[mi].u[1] = *(const uint4*)(p + 32);
    }
#pragma unroll
    for (int ni = 0; ni < 4; ++ni) {
      const char* p = bB + (nrow0 + ni * 16) * LDS_STRIDE + foB;
      fb[ni].u[0] = *(const uint4*)p;
      fb[ni].u[1] = *(const uint4*)(p + 32);
    }
#pragma unroll
    for (int mi = 0; mi < 2; ++mi)
#pragma unroll
      for (int ni = 0; ni < 4; ++ni)
        acc[mi][ni] = __builtin_amdgcn_wmma_f32_16x16x32_bf16(
            false, fa[mi].v, false, fb[ni].v, (short)0, acc[mi][ni], false, false);

    __syncthreads();            // done reading buf[cur]; next iter may refill it
  }
#else
  // -------- fallback: VGPR staging --------
  uint4 ra0, ra1, rb0, rb1;
  auto loadT = [&](int it, uint4& a0, uint4& a1, uint4& b0r, uint4& b1r) {
    const __bf16 *pa0, *pa1, *pb0, *pb1;
    srcAddrs(it, pa0, pa1, pb0, pb1);
    a0 = *(const uint4*)pa0;  a1 = *(const uint4*)pa1;
    b0r = *(const uint4*)pb0; b1r = *(const uint4*)pb1;
  };
  loadT(0, ra0, ra1, rb0, rb1);
  char* wA  = ldsA + srow * LDS_STRIDE + spiece * 16;
  char* wB  = ldsB + srow * LDS_STRIDE + spiece * 16;
  for (int it = 0; it < 144; ++it) {
    __syncthreads();
    *(uint4*)wA = ra0;  *(uint4*)(wA + 64 * LDS_STRIDE) = ra1;
    *(uint4*)wB = rb0;  *(uint4*)(wB + 64 * LDS_STRIDE) = rb1;
    __syncthreads();
    if (it + 1 < 144) loadT(it + 1, ra0, ra1, rb0, rb1);
    FragU fa[2], fb[4];
#pragma unroll
    for (int mi = 0; mi < 2; ++mi) {
      const char* p = ldsA + (mrow0 + mi * 16) * LDS_STRIDE + foA;
      fa[mi].u[0] = *(const uint4*)p;
      fa[mi].u[1] = *(const uint4*)(p + 32);
    }
#pragma unroll
    for (int ni = 0; ni < 4; ++ni) {
      const char* p = ldsB + (nrow0 + ni * 16) * LDS_STRIDE + foB;
      fb[ni].u[0] = *(const uint4*)p;
      fb[ni].u[1] = *(const uint4*)(p + 32);
    }
#pragma unroll
    for (int mi = 0; mi < 2; ++mi)
#pragma unroll
      for (int ni = 0; ni < 4; ++ni)
        acc[mi][ni] = __builtin_amdgcn_wmma_f32_16x16x32_bf16(
            false, fa[mi].v, false, fb[ni].v, (short)0, acc[mi][ni], false, false);
  }
#endif

  // Epilogue: per-(b,co) scale, then twice (+ns*noise + bias; leaky 0.2)
  const int mO = (lane >> 4) << 3;    // C/D layout: lanes>=16 hold M=r+8
#pragma unroll
  for (int ni = 0; ni < 4; ++ni) {
    int co = nblk * 128 + wn * 64 + ni * 16 + (lane & 15);
    float tc = t0[b * COUT + co];
    float nz = noise[co];
#pragma unroll
    for (int mi = 0; mi < 2; ++mi) {
#pragma unroll
      for (int r = 0; r < 8; ++r) {
        int m  = wm * 32 + mi * 16 + mO + r;
        int ry = 2 * y2 + (m >> 6);
        int px = m & 63;
        size_t idx = (((size_t)(b * H1 + ry)) * H1 + px) * COUT + co;
        float v = acc[mi][ni][r] * tc;
        v = v + ns0[idx] * nz + bias0[idx];
        v = (v > 0.f) ? v : 0.2f * v;
        v = v + ns1[idx] * nz + bias1[idx];
        v = (v > 0.f) ? v : 0.2f * v;
        out[idx] = v;
      }
    }
  }
}

// ---------------------------------------------------------------------------
extern "C" void kernel_launch(void* const* d_in, const int* in_sizes, int n_in,
                              void* d_out, int out_size, void* d_ws, size_t ws_size,
                              hipStream_t stream) {
  (void)in_sizes; (void)n_in; (void)out_size; (void)ws_size;
  const float* x     = (const float*)d_in[0];
  const float* w     = (const float*)d_in[1];
  const float* noise = (const float*)d_in[2];
  const float* y0k   = (const float*)d_in[3];
  const float* y0b   = (const float*)d_in[4];
  // d_in[5],[6],[8]: y1_kernel / y1_bias / conv1 -> dead code in reference
  const float* conv0 = (const float*)d_in[7];
  const float* ns0   = (const float*)d_in[9];
  const float* ns1   = (const float*)d_in[10];
  const float* b0    = (const float*)d_in[11];
  const float* b1    = (const float*)d_in[12];
  float* out = (float*)d_out;

  char* ws = (char*)d_ws;
  __bf16* xh = (__bf16*)(ws);                              // 35,684,352 B
  __bf16* wT = (__bf16*)(ws + 35684352);                   //  4,718,592 B
  float*  Q  = (float*)(ws + 35684352 + 4718592);          //      2,048 B
  float*  t0 = (float*)(ws + 35684352 + 4718592 + 2048);   //     16,384 B

  k_upsample<<<8712, 256, 0, stream>>>(x, xh);             // 8*66*66*64 threads
  k_wconv   <<<1152, 256, 0, stream>>>(conv0, wT);         // 9*512*64 threads
  k_q       <<<2,    256, 0, stream>>>(conv0, Q);
  k_style   <<<16,   256, 0, stream>>>(w, y0k, y0b, Q, t0);

  dim3 grid(4, 32, 8);                                     // (cout blk, row pair, batch)
  k_conv<<<grid, 256, 0, stream>>>(xh, wT, t0, noise, ns0, ns1, b0, b1, out);
}
